// Model_5454608466456
// MI455X (gfx1250) — compile-verified, matrix operand order checked by reading, not run on hardware
//
#include <hip/hip_runtime.h>
#include <hip/hip_bf16.h>

// ---------------------------------------------------------------------------
// CDNA5 (gfx1250) implementation: bf16 WMMA GEMMs (f32 accumulate) for all
// matrix work, fused GRU-gate epilogue, small pooling kernel for the tail.
// ---------------------------------------------------------------------------

typedef __attribute__((ext_vector_type(16))) __bf16 v16bf;
typedef __attribute__((ext_vector_type(8)))  float  v8f;

#define BDIM 512   // D
#define NNODE 256
#define NEDGE 512
#define NBATCH 64

// Generic tiled WMMA GEMM:
//   C[m,n] = epilogue( sum_k Ain[m,k] * W[k,n] + bias[n] )
// where Ain[m,k] = A0[m,k]                      for k <  Ksplit
//                = A1[m,k-Ksplit] (* A1mul[..]) for k >= Ksplit
// ACT: 0 = identity, 1 = sigmoid, 2 = tanh
// GATE: epilogue writes C[idx] = (1-Zg[idx])*PSold[idx] + Zg[idx]*act(v)
// Tile: BM=64, BN=64, BK=32; 8 waves; wave -> (m-tile = w&3, n-tiles = (w>>2)*2+{0,1})
template <int ACT, bool GATE>
__global__ __launch_bounds__(256) void wmma_gemm(
    const float* __restrict__ A0, long a0_batch, int lda0,
    const float* __restrict__ A1, int lda1,
    const float* __restrict__ A1mul, int ldm,
    const float* __restrict__ W, long w_batch, int ldw,
    const float* __restrict__ bias,
    float* __restrict__ C, long c_batch, int ldc,
    const float* __restrict__ Zg,
    const float* __restrict__ PSold,
    int Ksplit, int Ktotal)
{
    __shared__ __bf16 As[64][34];   // [row(m)][k]
    __shared__ __bf16 Bs[64][34];   // [col(n)][k] (transposed stage)

    const int tid  = threadIdx.x;
    const int lane = tid & 31;
    const int wave = tid >> 5;
    const int wm   = wave & 3;       // m-tile index within block tile
    const int wn2  = wave >> 2;      // n-tile pair index
    const int lr   = lane & 15;
    const int lh   = lane >> 4;

    const long a0b = (long)blockIdx.z * a0_batch;
    const long wb  = (long)blockIdx.z * w_batch;
    const long cb  = (long)blockIdx.z * c_batch;
    const int  m0  = blockIdx.y * 64;
    const int  n0  = blockIdx.x * 64;

    v8f acc0 = {};
    v8f acc1 = {};

    for (int kt = 0; kt < Ktotal; kt += 32) {
        // Stage A tile (64x32), f32 -> bf16. Consecutive tids walk K: coalesced.
        for (int i = tid; i < 2048; i += 256) {
            const int kk  = i & 31;
            const int row = i >> 5;
            const int gk  = kt + kk;
            float v;
            if (gk < Ksplit) {
                v = A0[a0b + (long)(m0 + row) * lda0 + gk];
            } else {
                const int k2 = gk - Ksplit;
                v = A1[(long)(m0 + row) * lda1 + k2];
                if (A1mul) v *= A1mul[(long)(m0 + row) * ldm + k2];
            }
            As[row][kk] = (__bf16)v;
        }
        // Stage B tile (32x64) transposed into Bs[n][k]. Consecutive tids walk N.
        for (int i = tid; i < 2048; i += 256) {
            const int n  = i & 63;
            const int kk = i >> 6;
            Bs[n][kk] = (__bf16)W[wb + (long)(kt + kk) * ldw + (n0 + n)];
        }
        __syncthreads();

        // Build fragments per ISA 16-bit 16x32 A layout:
        // lanes 0-15: VGPR0-3 -> K 0..7,  VGPR4-7 -> K 16..23
        // lanes16-31: VGPR0-3 -> K 8..15, VGPR4-7 -> K 24..31
        v16bf af, bf0, bf1;
        const int arow = wm * 16 + lr;
        const int b0row = (wn2 * 2 + 0) * 16 + lr;
        const int b1row = (wn2 * 2 + 1) * 16 + lr;
#pragma unroll
        for (int j = 0; j < 8; ++j) {
            const int kb = (j < 4) ? (lh * 8 + 2 * j) : (16 + lh * 8 + 2 * (j - 4));
            af[2 * j]      = As[arow][kb];
            af[2 * j + 1]  = As[arow][kb + 1];
            bf0[2 * j]     = Bs[b0row][kb];
            bf0[2 * j + 1] = Bs[b0row][kb + 1];
            bf1[2 * j]     = Bs[b1row][kb];
            bf1[2 * j + 1] = Bs[b1row][kb + 1];
        }
        acc0 = __builtin_amdgcn_wmma_f32_16x16x32_bf16(false, af, false, bf0,
                                                       (short)0, acc0, false, false);
        acc1 = __builtin_amdgcn_wmma_f32_16x16x32_bf16(false, af, false, bf1,
                                                       (short)0, acc1, false, false);
        __syncthreads();
    }

    // Epilogue. C fragment layout: VGPR r, lanes 0-15 -> M=r, lanes 16-31 -> M=8+r.
    const int nc0 = n0 + (wn2 * 2 + 0) * 16 + lr;
    const int nc1 = n0 + (wn2 * 2 + 1) * 16 + lr;
    const float bv0 = bias ? bias[nc0] : 0.0f;
    const float bv1 = bias ? bias[nc1] : 0.0f;
#pragma unroll
    for (int r = 0; r < 8; ++r) {
        const int mrow = m0 + wm * 16 + lh * 8 + r;
        {
            const long idx = cb + (long)mrow * ldc + nc0;
            float v = acc0[r] + bv0;
            if (ACT == 1) v = 1.0f / (1.0f + __expf(-v));
            else if (ACT == 2) v = tanhf(v);
            if (GATE) {
                const float zz = Zg[idx];
                const float p  = PSold[idx];
                C[idx] = (1.0f - zz) * p + zz * v;
            } else {
                C[idx] = v;
            }
        }
        {
            const long idx = cb + (long)mrow * ldc + nc1;
            float v = acc1[r] + bv1;
            if (ACT == 1) v = 1.0f / (1.0f + __expf(-v));
            else if (ACT == 2) v = tanhf(v);
            if (GATE) {
                const float zz = Zg[idx];
                const float p  = PSold[idx];
                C[idx] = (1.0f - zz) * p + zz * v;
            } else {
                C[idx] = v;
            }
        }
    }
}

// Attention pooling + residual tail. One block per batch element.
// t = tanh(ps@att1_W+b)  [B,N,D], o = tanh(ps@out_W+b) [B,N,D]
__global__ __launch_bounds__(256) void pool_kernel(
    const float* __restrict__ t, const float* __restrict__ o,
    const float* __restrict__ att2_W, const float* __restrict__ att2_b,
    const float* __restrict__ init_nodes, const float* __restrict__ init_edges,
    float* __restrict__ res, float* __restrict__ mul)
{
    const int b = blockIdx.x;
    const int tid = threadIdx.x;   // 256 == N
    __shared__ float red[256];
    __shared__ float wsh[256];

    // atten[b, tid] = dot(t[b,tid,:], att2_W) + att2_b
    const float* trow = t + ((long)b * NNODE + tid) * BDIM;
    float acc = 0.f;
    for (int d = 0; d < BDIM; ++d) acc += trow[d] * att2_W[d];
    acc += att2_b[0];

    // softmax over the 256 nodes
    red[tid] = acc;
    __syncthreads();
    for (int s = 128; s > 0; s >>= 1) {
        if (tid < s) red[tid] = fmaxf(red[tid], red[tid + s]);
        __syncthreads();
    }
    const float mx = red[0];
    __syncthreads();
    const float e = __expf(acc - mx);
    red[tid] = e;
    __syncthreads();
    for (int s = 128; s > 0; s >>= 1) {
        if (tid < s) red[tid] += red[tid + s];
        __syncthreads();
    }
    wsh[tid] = e / red[0];
    __syncthreads();

#pragma unroll
    for (int dd = 0; dd < 2; ++dd) {
        const int d = tid + dd * 256;
        float s = 0.f;
        for (int n = 0; n < NNODE; ++n)
            s += wsh[n] * o[((long)b * NNODE + n) * BDIM + d];
        mul[(long)b * BDIM + d] = s;                 // pre-tanh "mul" output
        const float g = tanhf(s);                    // graph_out
        float rsum = 0.f;
        for (int n = 0; n < NNODE; ++n)
            rsum += init_nodes[((long)b * NNODE + n) * BDIM + d];
        for (int e2 = 0; e2 < NEDGE; ++e2)
            rsum += init_edges[((long)b * NEDGE + e2) * BDIM + d];
        const float v = g + rsum * (1.0f / (float)(NNODE + NEDGE));
        res[(long)b * BDIM + d] = fmaxf(v, 0.f);
    }
}

extern "C" void kernel_launch(void* const* d_in, const int* in_sizes, int n_in,
                              void* d_out, int out_size, void* d_ws, size_t ws_size,
                              hipStream_t stream)
{
    const float* prop   = (const float*)d_in[0];   // [B,N,D]
    const float* edges  = (const float*)d_in[1];   // [B,E,D]
    const float* Amat   = (const float*)d_in[2];   // [B,N,E]
    // d_in[3] tem_features: unused by the reference computation
    const float* link_W = (const float*)d_in[4];
    const float* link_b = (const float*)d_in[5];
    const float* r_W    = (const float*)d_in[6];
    const float* r_b    = (const float*)d_in[7];
    const float* z_W    = (const float*)d_in[8];
    const float* z_b    = (const float*)d_in[9];
    const float* h_W    = (const float*)d_in[10];
    const float* h_b    = (const float*)d_in[11];
    const float* att1_W = (const float*)d_in[12];
    const float* att1_b = (const float*)d_in[13];
    const float* att2_W = (const float*)d_in[14];
    const float* att2_b = (const float*)d_in[15];
    const float* out_W  = (const float*)d_in[16];
    const float* out_b  = (const float*)d_in[17];

    const long PS_ELEMS = (long)NBATCH * NNODE * BDIM;   // 8,388,608
    const long ES_ELEMS = (long)NBATCH * NEDGE * BDIM;   // 16,777,216

    float* ws    = (float*)d_ws;
    float* es_ws = ws;                       // es ping buffer (16.7M f32)
    float* a_cur = es_ws + ES_ELEMS;         // bmm output / att1 activation
    float* psA   = a_cur + PS_ELEMS;         // ps ping
    float* psB   = psA + PS_ELEMS;           // ps pong
    float* rbuf  = psB + PS_ELEMS;           // reset gate / out activation
    float* zbuf  = rbuf + PS_ELEMS;          // update gate

    float* outp   = (float*)d_out;
    float* res    = outp;                            // [B,D]
    float* mulp   = outp + (long)NBATCH * BDIM;      // [B,1,D]
    float* es_out = mulp + (long)NBATCH * BDIM;      // [B,E,D]

    // seed ps with prop_state (capture-legal async D2D)
    hipMemcpyAsync(psA, prop, PS_ELEMS * sizeof(float),
                   hipMemcpyDeviceToDevice, stream);

    const dim3 blk(256);
    const dim3 grid_es(BDIM / 64, (NBATCH * NEDGE) / 64, 1);   // 8 x 512
    const dim3 grid_bmm(BDIM / 64, NNODE / 64, NBATCH);        // 8 x 4 x 64
    const dim3 grid_ps(BDIM / 64, (NBATCH * NNODE) / 64, 1);   // 8 x 256

    const float* es_in = edges;
    float* es_outs[5] = {es_out, es_ws, es_out, es_ws, es_out};
    float* ps_in = psA;
    float* ps_out = psB;

    for (int step = 0; step < 5; ++step) {
        float* es_o = es_outs[step];
        // es = es @ link_W + link_b        (M=32768, K=512, N=512)
        wmma_gemm<0, false><<<grid_es, blk, 0, stream>>>(
            es_in, 0L, BDIM, nullptr, 0, nullptr, 0,
            link_W, 0L, BDIM, link_b,
            es_o, 0L, BDIM, nullptr, nullptr, BDIM, BDIM);
        // a_cur[b] = A[b] @ es[b]          (batched, M=256, K=512, N=512)
        wmma_gemm<0, false><<<grid_bmm, blk, 0, stream>>>(
            Amat, (long)NNODE * NEDGE, NEDGE, nullptr, 0, nullptr, 0,
            es_o, (long)NEDGE * BDIM, BDIM, nullptr,
            a_cur, (long)NNODE * BDIM, BDIM, nullptr, nullptr, NEDGE, NEDGE);
        // r = sigmoid([a_cur, ps] @ r_W + r_b)
        wmma_gemm<1, false><<<grid_ps, blk, 0, stream>>>(
            a_cur, 0L, BDIM, ps_in, BDIM, nullptr, 0,
            r_W, 0L, BDIM, r_b,
            rbuf, 0L, BDIM, nullptr, nullptr, BDIM, 2 * BDIM);
        // z = sigmoid([a_cur, ps] @ z_W + z_b)
        wmma_gemm<1, false><<<grid_ps, blk, 0, stream>>>(
            a_cur, 0L, BDIM, ps_in, BDIM, nullptr, 0,
            z_W, 0L, BDIM, z_b,
            zbuf, 0L, BDIM, nullptr, nullptr, BDIM, 2 * BDIM);
        // ps_new = (1-z)*ps + z*tanh([a_cur, r*ps] @ h_W + h_b)   (fused)
        wmma_gemm<2, true><<<grid_ps, blk, 0, stream>>>(
            a_cur, 0L, BDIM, ps_in, BDIM, rbuf, BDIM,
            h_W, 0L, BDIM, h_b,
            ps_out, 0L, BDIM, zbuf, ps_in, BDIM, 2 * BDIM);

        es_in = es_o;
        float* tmp = ps_in; ps_in = ps_out; ps_out = tmp;
    }
    // after 5 steps, final ps is in ps_in; final es already in d_out es region.

    // t = tanh(ps @ att1_W + att1_b)   -> reuse a_cur
    wmma_gemm<2, false><<<grid_ps, blk, 0, stream>>>(
        ps_in, 0L, BDIM, nullptr, 0, nullptr, 0,
        att1_W, 0L, BDIM, att1_b,
        a_cur, 0L, BDIM, nullptr, nullptr, BDIM, BDIM);
    // o = tanh(ps @ out_W + out_b)     -> reuse rbuf
    wmma_gemm<2, false><<<grid_ps, blk, 0, stream>>>(
        ps_in, 0L, BDIM, nullptr, 0, nullptr, 0,
        out_W, 0L, BDIM, out_b,
        rbuf, 0L, BDIM, nullptr, nullptr, BDIM, BDIM);

    // attention softmax pooling + residual + relu
    pool_kernel<<<NBATCH, blk, 0, stream>>>(
        a_cur, rbuf, att2_W, att2_b, prop, edges, res, mulp);
}
